// SelfAttentionLayer_60748017435243
// MI455X (gfx1250) — matrix operand order repaired and checked by reading, not compiled, not run on hardware
//
#include <hip/hip_runtime.h>
#include <hip/hip_bf16.h>

// ---------------------------------------------------------------------------
// Self-attention layer for MI455X (gfx1250), bf16 WMMA + flash softmax +
// async global->LDS double-buffered K/V staging (ASYNCcnt path).
//   x:(1,4096,1024) f32, W*:(1024,1024) f32, b*:(1024) f32 -> out (1,4096,1024) f32
// ---------------------------------------------------------------------------

typedef __attribute__((ext_vector_type(16))) __bf16 v16bf;
typedef __attribute__((ext_vector_type(8)))  __bf16 v8bf;
typedef __attribute__((ext_vector_type(8)))  float  v8f;

#define S_LEN 4096
#define EMB   1024
#define NH    16
#define DH    64

#define WMMA_BF16(A, B, C) \
  __builtin_amdgcn_wmma_f32_16x16x32_bf16(false, (A), false, (B), (short)0, (C), false, false)

static __device__ __forceinline__ v8f zero8() {
  v8f z = {0.f, 0.f, 0.f, 0.f, 0.f, 0.f, 0.f, 0.f};
  return z;
}

static __device__ __forceinline__ v16bf cat8(v8bf lo, v8bf hi) {
  return __builtin_shufflevector(lo, hi, 0,1,2,3,4,5,6,7,8,9,10,11,12,13,14,15);
}

// A-fragment: 16x32 bf16 tile of a row-major matrix (leading dim = ld).
// Per CDNA5 ISA: lane m = lane%16; elements 0..7 = K 8*(lane/16)+0..7,
// elements 8..15 = K 16+8*(lane/16)+0..7. Both chunks are 16B contiguous.
static __device__ __forceinline__ v16bf frag_a(const __bf16* base, int row0, int k0,
                                               int ld, int lane) {
  const int m  = row0 + (lane & 15);
  const int kb = k0 + ((lane >> 4) << 3);
  const __bf16* p = base + (size_t)m * ld + kb;
  v8bf lo = *(const v8bf*)p;
  v8bf hi = *(const v8bf*)(p + 16);
  return cat8(lo, hi);
}

// B-fragment: 32x16 bf16 where column n is contiguous along k in memory,
// element (k, n) at base[n*ld + k]. Per CDNA5 ISA: lane column n = lane%16,
// elements 0..7 = K 8*(lane/16)+0..7, elements 8..15 = K 16+8*(lane/16)+0..7.
static __device__ __forceinline__ v16bf frag_b(const __bf16* base, int n0, int k0,
                                               int ld, int lane) {
  const int n  = n0 + (lane & 15);
  const int kb = k0 + ((lane >> 4) << 3);
  const __bf16* p = base + (size_t)n * ld + kb;
  v8bf lo = *(const v8bf*)p;
  v8bf hi = *(const v8bf*)(p + 16);
  return cat8(lo, hi);
}

// Async 16B global -> LDS copy (gfx1250, tracked by ASYNCcnt).
// lds_off = wave-relative LDS byte address (low 32 bits of the generic pointer).
static __device__ __forceinline__ void async_b128(unsigned lds_off,
                                                  const __bf16* gptr) {
  asm volatile("global_load_async_to_lds_b128 %0, %1, off"
               :: "v"(lds_off), "v"(gptr) : "memory");
}

// ---------------------------------------------------------------------------
// f32 -> bf16 conversion
// ---------------------------------------------------------------------------
__global__ __launch_bounds__(256) void cvt_bf16_kernel(const float* __restrict__ src,
                                                       __bf16* __restrict__ dst, int n) {
  int i = blockIdx.x * blockDim.x + threadIdx.x;
  int stride = gridDim.x * blockDim.x;
  for (; i < n; i += stride) dst[i] = (__bf16)src[i];
}

// ---------------------------------------------------------------------------
// QKV projection: dst = x @ W^T + b   (W row-major [out][in] => B column n = W row n)
// vmode 0: dst is [NH][S][DH] bf16 (Q / K storage)
// vmode 1: dst is [NH][DH][S] bf16 (V stored transposed for the P@V B-fragments)
// ---------------------------------------------------------------------------
__global__ __launch_bounds__(256) void qkv_gemm_kernel(const __bf16* __restrict__ xb,
                                                       const __bf16* __restrict__ Wb,
                                                       const float* __restrict__ bias,
                                                       __bf16* __restrict__ dst, int vmode) {
  const int lane = threadIdx.x & 31;
  const int wave = threadIdx.x >> 5;
  const int row0 = (blockIdx.y * 8 + wave) * 16;
  const int n0   = blockIdx.x * 64;

  v8f acc[4];
#pragma unroll
  for (int t = 0; t < 4; ++t) acc[t] = zero8();

  for (int k0 = 0; k0 < EMB; k0 += 32) {
    __builtin_prefetch(Wb + (size_t)(n0 + (lane & 15)) * EMB + k0 + 128, 0, 1);
    v16bf a = frag_a(xb, row0, k0, EMB, lane);
#pragma unroll
    for (int t = 0; t < 4; ++t)
      acc[t] = WMMA_BF16(a, frag_b(Wb, n0 + 16 * t, k0, EMB, lane), acc[t]);
  }

  const int hi = lane >> 4;
  const int nl = lane & 15;
#pragma unroll
  for (int t = 0; t < 4; ++t) {
    const int n = n0 + 16 * t + nl;
    const float bv = bias[n];
    const int h = n >> 6;   // head
    const int d = n & 63;   // dim within head
    if (vmode == 0) {
#pragma unroll
      for (int r = 0; r < 8; ++r) {
        const int m = row0 + r + 8 * hi;
        dst[((size_t)h * S_LEN + m) * DH + d] = (__bf16)(acc[t][r] + bv);
      }
    } else {
      const size_t b = ((size_t)h * DH + d) * S_LEN + row0 + 8 * hi;
      v8bf o;
#pragma unroll
      for (int r = 0; r < 8; ++r) o[r] = (__bf16)(acc[t][r] + bv);
      *(v8bf*)(dst + b) = o;  // 8 consecutive tokens, 16B store
    }
  }
}

// ---------------------------------------------------------------------------
// Flash attention. Block = 8 waves, all on the same head; each wave owns 16
// query rows. Per 32-key tile, the block cooperatively stages K (32x64) and
// V^T (64x32) into LDS with async global->LDS B128 copies, double-buffered:
// issue tile i+1, s_wait_asynccnt 2 (tile i landed), barrier, compute, barrier.
// ---------------------------------------------------------------------------
__global__ __launch_bounds__(256) void attn_kernel(const __bf16* __restrict__ Qh,
                                                   const __bf16* __restrict__ Kh,
                                                   const __bf16* __restrict__ Vt,
                                                   __bf16* __restrict__ AO) {
  // smem[b][0..2047]    : K tile, [32 keys][64 dims] row-major
  // smem[b][2048..4095] : V tile, [64 dims][32 keys] row-major
  __shared__ __bf16 smem[2][4096];
  __shared__ __bf16 ptile[8][16 * 32];

  const int tid  = threadIdx.x;
  const int lane = tid & 31;
  const int wave = tid >> 5;
  const int hi   = lane >> 4;
  const int nl   = lane & 15;
  const int head = blockIdx.y;
  const int row0 = (blockIdx.x * 8 + wave) * 16;
  const float SCALE = 0.125f;  // 1/sqrt(64)

  const __bf16* Qb = Qh + (size_t)head * S_LEN * DH;  // [S][DH]
  const __bf16* Kb = Kh + (size_t)head * S_LEN * DH;  // [S][DH]
  const __bf16* Vb = Vt + (size_t)head * DH * S_LEN;  // [DH][S]

  // per-thread async-copy slots (one B128 for K, one for V, per tile)
  const int vd = tid >> 2;          // dim row for V tile
  const int vq = tid & 3;           // 8-element chunk within the 32-key row
  const unsigned k_lds0 = (unsigned)(size_t)&smem[0][tid * 8];
  const unsigned v_lds0 = (unsigned)(size_t)&smem[0][2048 + vd * 32 + vq * 8];
  const unsigned k_lds1 = (unsigned)(size_t)&smem[1][tid * 8];
  const unsigned v_lds1 = (unsigned)(size_t)&smem[1][2048 + vd * 32 + vq * 8];

  // Q fragments persist across the whole key loop (Dh=64 -> two K=32 frags)
  const v16bf qf0 = frag_a(Qb, row0, 0,  DH, lane);
  const v16bf qf1 = frag_a(Qb, row0, 32, DH, lane);

  v8f ov[4];
#pragma unroll
  for (int t = 0; t < 4; ++t) ov[t] = zero8();
  float rmax[8], rsum[8];
#pragma unroll
  for (int r = 0; r < 8; ++r) { rmax[r] = -1e30f; rsum[r] = 0.f; }

  const int NT = S_LEN / 32;  // 128 key tiles

  // prologue: stage tile 0 into buffer 0
  async_b128(k_lds0, Kb + tid * 8);                    // K rows 0..31 contiguous
  async_b128(v_lds0, Vb + (size_t)vd * S_LEN + vq * 8);

  for (int it = 0; it < NT; ++it) {
    const int t0 = it * 32;
    // issue next tile into the other buffer, then wait for current tile
    if (it + 1 < NT) {
      const int tn = t0 + 32;
      const unsigned kl = ((it + 1) & 1) ? k_lds1 : k_lds0;
      const unsigned vl = ((it + 1) & 1) ? v_lds1 : v_lds0;
      async_b128(kl, Kb + (size_t)tn * DH + tid * 8);
      async_b128(vl, Vb + (size_t)vd * S_LEN + tn + vq * 8);
      asm volatile("s_wait_asynccnt 0x2" ::: "memory");
    } else {
      asm volatile("s_wait_asynccnt 0x0" ::: "memory");
    }
    __syncthreads();  // tile `it` visible to all waves

    const __bf16* ktile = &smem[it & 1][0];     // [32][64]
    const __bf16* vtile = &smem[it & 1][2048];  // [64][32]

    // scores: Q(16x64) @ K^T(64x32); K^T column = key row of the LDS tile
    v8f s0 = WMMA_BF16(qf0, frag_b(ktile, 0,  0,  DH, lane), zero8());
    s0     = WMMA_BF16(qf1, frag_b(ktile, 0,  32, DH, lane), s0);
    v8f s1 = WMMA_BF16(qf0, frag_b(ktile, 16, 0,  DH, lane), zero8());
    s1     = WMMA_BF16(qf1, frag_b(ktile, 16, 32, DH, lane), s1);

    // online softmax: each 16-lane half holds rows r+8*hi, one column per lane
    float p0[8], p1[8], al[8];
#pragma unroll
    for (int r = 0; r < 8; ++r) {
      float a = s0[r] * SCALE;
      float b = s1[r] * SCALE;
      float mx = fmaxf(a, b);
      mx = fmaxf(mx, __shfl_xor(mx, 1, 32));
      mx = fmaxf(mx, __shfl_xor(mx, 2, 32));
      mx = fmaxf(mx, __shfl_xor(mx, 4, 32));
      mx = fmaxf(mx, __shfl_xor(mx, 8, 32));
      const float nm = fmaxf(rmax[r], mx);
      const float alpha = __expf(rmax[r] - nm);
      rmax[r] = nm;
      const float e0 = __expf(a - nm);
      const float e1 = __expf(b - nm);
      p0[r] = e0; p1[r] = e1;
      float ps = e0 + e1;
      ps += __shfl_xor(ps, 1, 32);
      ps += __shfl_xor(ps, 2, 32);
      ps += __shfl_xor(ps, 4, 32);
      ps += __shfl_xor(ps, 8, 32);
      rsum[r] = rsum[r] * alpha + ps;
      al[r] = alpha;
    }
#pragma unroll
    for (int t = 0; t < 4; ++t)
#pragma unroll
      for (int r = 0; r < 8; ++r) ov[t][r] *= al[r];

    // transpose P from C-layout to A-layout via per-wave LDS tile
    __bf16* pt = &ptile[wave][0];
#pragma unroll
    for (int r = 0; r < 8; ++r) {
      const int m = r + 8 * hi;
      pt[m * 32 + nl]      = (__bf16)p0[r];
      pt[m * 32 + nl + 16] = (__bf16)p1[r];
    }
    asm volatile("s_wait_dscnt 0" ::: "memory");
    const v16bf pf = frag_a(pt, 0, 0, 32, lane);

    // O += P(16x32) @ V(32x64): V tile row d is contiguous along keys
#pragma unroll
    for (int t = 0; t < 4; ++t)
      ov[t] = WMMA_BF16(pf, frag_b(vtile, 16 * t, 0, 32, lane), ov[t]);

    __syncthreads();  // all waves done reading this buffer
  }

  // normalize and write concat-head activation AO[S][EMB] in bf16
#pragma unroll
  for (int t = 0; t < 4; ++t) {
    const int col = head * DH + 16 * t + nl;
#pragma unroll
    for (int r = 0; r < 8; ++r) {
      const int m = row0 + r + 8 * hi;
      AO[(size_t)m * EMB + col] = (__bf16)(ov[t][r] / rsum[r]);
    }
  }
}

// ---------------------------------------------------------------------------
// Output projection: out = AO @ Wo^T + bo, f32 result
// ---------------------------------------------------------------------------
__global__ __launch_bounds__(256) void out_gemm_kernel(const __bf16* __restrict__ AO,
                                                       const __bf16* __restrict__ Wb,
                                                       const float* __restrict__ bias,
                                                       float* __restrict__ out) {
  const int lane = threadIdx.x & 31;
  const int wave = threadIdx.x >> 5;
  const int row0 = (blockIdx.y * 8 + wave) * 16;
  const int n0   = blockIdx.x * 64;

  v8f acc[4];
#pragma unroll
  for (int t = 0; t < 4; ++t) acc[t] = zero8();

  for (int k0 = 0; k0 < EMB; k0 += 32) {
    __builtin_prefetch(Wb + (size_t)(n0 + (lane & 15)) * EMB + k0 + 128, 0, 1);
    v16bf a = frag_a(AO, row0, k0, EMB, lane);
#pragma unroll
    for (int t = 0; t < 4; ++t)
      acc[t] = WMMA_BF16(a, frag_b(Wb, n0 + 16 * t, k0, EMB, lane), acc[t]);
  }

  const int hi = lane >> 4;
  const int nl = lane & 15;
#pragma unroll
  for (int t = 0; t < 4; ++t) {
    const int n = n0 + 16 * t + nl;
    const float bv = bias[n];
#pragma unroll
    for (int r = 0; r < 8; ++r) {
      const int m = row0 + r + 8 * hi;
      out[(size_t)m * EMB + n] = acc[t][r] + bv;
    }
  }
}

// ---------------------------------------------------------------------------
// Host launcher
// ---------------------------------------------------------------------------
extern "C" void kernel_launch(void* const* d_in, const int* in_sizes, int n_in,
                              void* d_out, int out_size, void* d_ws, size_t ws_size,
                              hipStream_t stream) {
  (void)in_sizes; (void)n_in; (void)out_size; (void)ws_size;

  const float* x  = (const float*)d_in[0];
  const float* Wq = (const float*)d_in[1];
  const float* bq = (const float*)d_in[2];
  const float* Wk = (const float*)d_in[3];
  const float* bk = (const float*)d_in[4];
  const float* Wv = (const float*)d_in[5];
  const float* bv = (const float*)d_in[6];
  const float* Wo = (const float*)d_in[7];
  const float* bo = (const float*)d_in[8];
  float* out = (float*)d_out;

  // Workspace layout (bf16 elements), total 24M elems = 48 MB
  __bf16* xb  = (__bf16*)d_ws;                    // [S][EMB]
  __bf16* Wqb = xb  + (size_t)S_LEN * EMB;        // [EMB][EMB]
  __bf16* Wkb = Wqb + (size_t)EMB * EMB;
  __bf16* Wvb = Wkb + (size_t)EMB * EMB;
  __bf16* Wob = Wvb + (size_t)EMB * EMB;
  __bf16* Qh  = Wob + (size_t)EMB * EMB;          // [NH][S][DH]
  __bf16* Kh  = Qh  + (size_t)S_LEN * EMB;        // [NH][S][DH]
  __bf16* Vt  = Kh  + (size_t)S_LEN * EMB;        // [NH][DH][S]
  __bf16* AO  = Vt  + (size_t)S_LEN * EMB;        // [S][EMB]

  cvt_bf16_kernel<<<2048, 256, 0, stream>>>(x,  xb,  S_LEN * EMB);
  cvt_bf16_kernel<<<1024, 256, 0, stream>>>(Wq, Wqb, EMB * EMB);
  cvt_bf16_kernel<<<1024, 256, 0, stream>>>(Wk, Wkb, EMB * EMB);
  cvt_bf16_kernel<<<1024, 256, 0, stream>>>(Wv, Wvb, EMB * EMB);
  cvt_bf16_kernel<<<1024, 256, 0, stream>>>(Wo, Wob, EMB * EMB);

  dim3 gg(EMB / 64, S_LEN / 128);  // (16, 32)
  qkv_gemm_kernel<<<gg, 256, 0, stream>>>(xb, Wqb, bq, Qh, 0);
  qkv_gemm_kernel<<<gg, 256, 0, stream>>>(xb, Wkb, bk, Kh, 0);
  qkv_gemm_kernel<<<gg, 256, 0, stream>>>(xb, Wvb, bv, Vt, 1);

  attn_kernel<<<dim3(S_LEN / 128, NH), 256, 0, stream>>>(Qh, Kh, Vt, AO);

  out_gemm_kernel<<<gg, 256, 0, stream>>>(AO, Wob, bo, out);
}
